// SceneIdentifier_77429670412476
// MI455X (gfx1250) — matrix operand order repaired and checked by reading, not compiled, not run on hardware
//
#include <hip/hip_runtime.h>
#include <hip/hip_bf16.h>
#include <stdint.h>

// ---------------------------------------------------------------------------
// CDNA5 (gfx1250) fused SceneIdentifier:
//   relu(X@W1+b1) -> relu(@W2+b2) -> @W3+b3 -> row-normalize -> @embn^T -> argmax
// All GEMMs via v_wmma_f32_16x16x32_bf16 (wave32). X staged to LDS as bf16
// once per block. All LDS tiles use bank-rotated (padded) row strides so the
// 16B A-fragment loads are conflict-free across the 64 LDS banks.
// ---------------------------------------------------------------------------

typedef __attribute__((ext_vector_type(16))) __bf16 bf16x16;
typedef __attribute__((ext_vector_type(2)))  __bf16 bf16x2;
typedef __attribute__((ext_vector_type(8)))  float  v8f;

union ABf {
  bf16x16 v;
  uint32_t u[8];
  uint4    q[2];
};

static __device__ __forceinline__ uint32_t pk2(float lo, float hi) {
#if defined(__gfx1250__) && __has_builtin(__builtin_amdgcn_cvt_pk_bf16_f32)
  union { bf16x2 v; uint32_t u; } cv;
  cv.v = __builtin_amdgcn_cvt_pk_bf16_f32(lo, hi);   // v_cvt_pk_bf16_f32
  return cv.u;
#else
  uint32_t a = __float_as_uint(lo), b = __float_as_uint(hi);
  a += 0x7FFFu + ((a >> 16) & 1u);                   // RNE
  b += 0x7FFFu + ((b >> 16) & 1u);
  return (a >> 16) | (b & 0xFFFF0000u);
#endif
}
static __device__ __forceinline__ uint16_t f2bf(float f) {
  return (uint16_t)pk2(f, 0.f);
}
static __device__ __forceinline__ v8f wmma_bf16(ABf a, ABf b, v8f c) {
  // 8 args: (neg_a, A, neg_b, B, c_mod, C, reuse_a, reuse_b)
  return __builtin_amdgcn_wmma_f32_16x16x32_bf16(false, a.v, false, b.v,
                                                 (short)0, c, false, false);
}

#define D0 1024
#define D1 512
#define D2 256
#define D3 128
#define NE 1000
#define MROWS 32   // batch rows per block

// Padded LDS row strides (elements). bf16 strides are == 16B (mod 256B) so
// 16 consecutive rows start in 16 disjoint 4-bank groups (64 banks covered).
#define XS   1032  // Xb row stride (bf16): 2064B
#define HS1  520   // H1 row stride (bf16): 1040B
#define HS2  264   // H2 row stride (bf16): 528B
#define PS   136   // Pn row stride (bf16): 272B
#define PFS  129   // Pf row stride (f32):  516B  (== 1 bank rotation)
#define RS   65    // Rv/Ri row stride (4B entries, rotates banks)

// ---------------- pre-pass: transpose weights to bf16 [N][K] ----------------
__global__ __launch_bounds__(256) void prep_weights(
    const float* __restrict__ W1, const float* __restrict__ W2,
    const float* __restrict__ W3,
    uint16_t* __restrict__ W1t, uint16_t* __restrict__ W2t,
    uint16_t* __restrict__ W3t) {
  int i = blockIdx.x * 256 + threadIdx.x;
  if (i < 524288) {                       // W1t[512][1024]
    int n = i >> 10, k = i & 1023;
    W1t[i] = f2bf(W1[k * D1 + n]);
  } else if (i < 655360) {                // W2t[256][512]
    int j = i - 524288;
    int n = j >> 9, k = j & 511;
    W2t[j] = f2bf(W2[k * D2 + n]);
  } else if (i < 688128) {                // W3t[128][256]
    int j = i - 655360;
    int n = j >> 8, k = j & 255;
    W3t[j] = f2bf(W3[k * D3 + n]);
  }
}

// ------------- pre-pass: normalize embeddings, pad rows to 1024 -------------
__global__ __launch_bounds__(256) void prep_emb(const float* __restrict__ emb,
                                                uint16_t* __restrict__ EMB) {
  int row = blockIdx.x * 256 + threadIdx.x;
  if (row >= 1024) return;
  if (row >= NE) {
    for (int j = 0; j < D3; ++j) EMB[row * D3 + j] = 0;
    return;
  }
  float ss = 0.f;
  for (int j = 0; j < D3; ++j) { float v = emb[row * D3 + j]; ss += v * v; }
  float s = 1.f / fmaxf(sqrtf(ss), 1e-8f);
  for (int j = 0; j < D3; ++j) EMB[row * D3 + j] = f2bf(emb[row * D3 + j] * s);
}

// -------------------------------- main kernel -------------------------------
__global__ __launch_bounds__(128) void scene_main(
    const float* __restrict__ X,
    const float* __restrict__ b1v, const float* __restrict__ b2v,
    const float* __restrict__ b3v,
    const uint16_t* __restrict__ W1t, const uint16_t* __restrict__ W2t,
    const uint16_t* __restrict__ W3t, const uint16_t* __restrict__ EMB,
    int* __restrict__ out) {
  // Region 0: Xb [0, 66048) + H1 [66048, 99328) live through layer 2.
  // The Xb region is re-carved after layer 1 (barrier-protected).
  __shared__ __align__(16) unsigned char smem[99328];
  uint16_t* Xb = (uint16_t*)smem;              // [32][XS]  bf16
  uint16_t* H1 = (uint16_t*)(smem + 66048);    // [32][HS1] bf16
  uint16_t* H2 = (uint16_t*)smem;              // [32][HS2] bf16 [0,16896)
  float*    Pf = (float*)(smem + 16896);       // [32][PFS] f32  [16896,33408)
  uint16_t* Pn = (uint16_t*)(smem + 33408);    // [32][PS]  bf16 [33408,42112)
  float*    Rv = (float*)(smem + 42112);       // [32][RS]       [42112,50432)
  int*      Ri = (int*)(smem + 50432);         // [32][RS]       [50432,58752)

  const int tid  = threadIdx.x;
  const int w    = tid >> 5;        // wave 0..3, owns 1/4 of N everywhere
  const int lane = tid & 31;
  const int l16  = lane & 15;
  const int half = lane >> 4;
  const int rofs = half << 3;       // C/D: upper 16 lanes hold rows M+8
  const int row0 = blockIdx.x * MROWS;

  // ============ stage X tile to LDS as bf16 (read X from HBM once) ==========
  {
    const float4* xs = (const float4*)(X + (size_t)row0 * D0);
    #pragma unroll 8
    for (int c = 0; c < 64; ++c) {
      int idx = c * 128 + tid;           // float4 index in tile, coalesced
      float4 f = xs[idx];
      uint2 p;
      p.x = pk2(f.x, f.y);
      p.y = pk2(f.z, f.w);
      int row = idx >> 8, c4 = idx & 255;
      *(uint2*)(Xb + row * XS + c4 * 4) = p;   // ds_store_b64
    }
  }
  __syncthreads();

  // ================= layer 1: Xb[32x1024] @ W1 -> H1[32x512] ================
  for (int g = 0; g < 2; ++g) {
    v8f acc[2][4];
    #pragma unroll
    for (int t = 0; t < 2; ++t)
      #pragma unroll
      for (int j = 0; j < 4; ++j) { v8f z = {}; acc[t][j] = z; }

    for (int k = 0; k < 32; ++k) {
      ABf a[2];
      #pragma unroll
      for (int t = 0; t < 2; ++t) {
        // A frag 16x32 bf16: lanes 0-15 K[0-7],[16-23]; lanes 16-31 +8
        const uint16_t* hp = Xb + (t * 16 + l16) * XS + k * 32 + half * 8;
        a[t].q[0] = *(const uint4*)hp;
        a[t].q[1] = *(const uint4*)(hp + 16);
      }
      #pragma unroll
      for (int j = 0; j < 4; ++j) {
        int n0 = w * 128 + g * 64 + j * 16;
        const uint16_t* bp =
            W1t + (size_t)(n0 + l16) * D0 + k * 32 + half * 8;
        ABf b;
        b.q[0] = *(const uint4*)bp;
        b.q[1] = *(const uint4*)(bp + 16);
        acc[0][j] = wmma_bf16(a[0], b, acc[0][j]);
        acc[1][j] = wmma_bf16(a[1], b, acc[1][j]);
      }
    }
    #pragma unroll
    for (int j = 0; j < 4; ++j) {
      int col = w * 128 + g * 64 + j * 16 + l16;
      float bias = b1v[col];
      #pragma unroll
      for (int t = 0; t < 2; ++t)
        #pragma unroll
        for (int r = 0; r < 8; ++r) {
          int row = t * 16 + r + rofs;
          H1[row * HS1 + col] = f2bf(fmaxf(acc[t][j][r] + bias, 0.f));
        }
    }
  }
  __syncthreads();

  // ================= layer 2: H1[32x512] @ W2 -> H2[32x256] =================
  {
    v8f acc[2][4];
    #pragma unroll
    for (int t = 0; t < 2; ++t)
      #pragma unroll
      for (int j = 0; j < 4; ++j) { v8f z = {}; acc[t][j] = z; }

    for (int k = 0; k < 16; ++k) {
      ABf a[2];
      #pragma unroll
      for (int t = 0; t < 2; ++t) {
        const uint16_t* hp = H1 + (t * 16 + l16) * HS1 + k * 32 + half * 8;
        a[t].q[0] = *(const uint4*)hp;
        a[t].q[1] = *(const uint4*)(hp + 16);
      }
      #pragma unroll
      for (int j = 0; j < 4; ++j) {
        int n0 = w * 64 + j * 16;
        const uint16_t* bp =
            W2t + (size_t)(n0 + l16) * D1 + k * 32 + half * 8;
        ABf b;
        b.q[0] = *(const uint4*)bp;
        b.q[1] = *(const uint4*)(bp + 16);
        acc[0][j] = wmma_bf16(a[0], b, acc[0][j]);
        acc[1][j] = wmma_bf16(a[1], b, acc[1][j]);
      }
    }
    #pragma unroll
    for (int j = 0; j < 4; ++j) {
      int col = w * 64 + j * 16 + l16;
      float bias = b2v[col];
      #pragma unroll
      for (int t = 0; t < 2; ++t)
        #pragma unroll
        for (int r = 0; r < 8; ++r) {
          int row = t * 16 + r + rofs;
          H2[row * HS2 + col] = f2bf(fmaxf(acc[t][j][r] + bias, 0.f));
        }
    }
  }
  __syncthreads();

  // ================= layer 3: H2[32x256] @ W3 -> Pf[32x128] =================
  {
    v8f acc[2][2];
    #pragma unroll
    for (int t = 0; t < 2; ++t)
      #pragma unroll
      for (int j = 0; j < 2; ++j) { v8f z = {}; acc[t][j] = z; }

    for (int k = 0; k < 8; ++k) {
      ABf a[2];
      #pragma unroll
      for (int t = 0; t < 2; ++t) {
        const uint16_t* hp = H2 + (t * 16 + l16) * HS2 + k * 32 + half * 8;
        a[t].q[0] = *(const uint4*)hp;
        a[t].q[1] = *(const uint4*)(hp + 16);
      }
      #pragma unroll
      for (int j = 0; j < 2; ++j) {
        int n0 = w * 32 + j * 16;
        const uint16_t* bp =
            W3t + (size_t)(n0 + l16) * D2 + k * 32 + half * 8;
        ABf b;
        b.q[0] = *(const uint4*)bp;
        b.q[1] = *(const uint4*)(bp + 16);
        acc[0][j] = wmma_bf16(a[0], b, acc[0][j]);
        acc[1][j] = wmma_bf16(a[1], b, acc[1][j]);
      }
    }
    #pragma unroll
    for (int j = 0; j < 2; ++j) {
      int col = w * 32 + j * 16 + l16;
      float bias = b3v[col];
      #pragma unroll
      for (int t = 0; t < 2; ++t)
        #pragma unroll
        for (int r = 0; r < 8; ++r) {
          int row = t * 16 + r + rofs;
          Pf[row * PFS + col] = acc[t][j][r] + bias;
        }
    }
  }
  __syncthreads();

  // ============ row-normalize Pf -> Pn (bf16), torch cosine eps =============
  if (tid < 32) {
    int row = tid;
    float ss = 0.f;
    for (int j = 0; j < D3; ++j) { float v = Pf[row * PFS + j]; ss += v * v; }
    float s = 1.f / fmaxf(sqrtf(ss), 1e-8f);
    for (int j = 0; j < D3; ++j) Pn[row * PS + j] = f2bf(Pf[row * PFS + j] * s);
  }
  __syncthreads();

  // ========= sims = Pn[32x128] @ EMB^T[128x1024], fused lane argmax =========
  {
    float bv[16];
    int   bi[16];
    #pragma unroll
    for (int i = 0; i < 16; ++i) { bv[i] = -3.4e38f; bi[i] = 0x7FFFFFFF; }

    for (int nt = 0; nt < 16; ++nt) {
      int n0 = w * 256 + nt * 16;
      v8f acc[2];
      { v8f z = {}; acc[0] = z; acc[1] = z; }
      #pragma unroll
      for (int k = 0; k < 4; ++k) {
        ABf a[2];
        #pragma unroll
        for (int t = 0; t < 2; ++t) {
          const uint16_t* pp = Pn + (t * 16 + l16) * PS + k * 32 + half * 8;
          a[t].q[0] = *(const uint4*)pp;
          a[t].q[1] = *(const uint4*)(pp + 16);
        }
        const uint16_t* bp =
            EMB + (size_t)(n0 + l16) * D3 + k * 32 + half * 8;
        ABf b;
        b.q[0] = *(const uint4*)bp;
        b.q[1] = *(const uint4*)(bp + 16);
        acc[0] = wmma_bf16(a[0], b, acc[0]);
        acc[1] = wmma_bf16(a[1], b, acc[1]);
      }
      int col = n0 + l16;                   // C/D layout: lane -> N column
      if (col < NE) {
        #pragma unroll
        for (int t = 0; t < 2; ++t)
          #pragma unroll
          for (int r = 0; r < 8; ++r) {
            float v = acc[t][r];
            int s = t * 8 + r;
            if (v > bv[s]) { bv[s] = v; bi[s] = col; }
          }
      }
    }
    int slot = w * 16 + l16;
    #pragma unroll
    for (int t = 0; t < 2; ++t)
      #pragma unroll
      for (int r = 0; r < 8; ++r) {
        int row = t * 16 + r + rofs;
        Rv[row * RS + slot] = bv[t * 8 + r];
        Ri[row * RS + slot] = bi[t * 8 + r];
      }
  }
  __syncthreads();

  // ================== final cross-lane/wave argmax reduce ===================
  if (tid < 32) {
    int row = tid;
    float best = -3.4e38f;
    int bidx = 0x7FFFFFFF;
    for (int s = 0; s < 64; ++s) {
      float v = Rv[row * RS + s];
      int i = Ri[row * RS + s];
      if (v > best || (v == best && i < bidx)) { best = v; bidx = i; }
    }
    out[row0 + row] = bidx;
  }
}

// --------------------------------- launcher ---------------------------------
extern "C" void kernel_launch(void* const* d_in, const int* in_sizes, int n_in,
                              void* d_out, int out_size, void* d_ws,
                              size_t ws_size, hipStream_t stream) {
  (void)in_sizes; (void)n_in; (void)out_size; (void)ws_size;
  const float* X   = (const float*)d_in[0];
  const float* W1  = (const float*)d_in[1];
  const float* b1  = (const float*)d_in[2];
  const float* W2  = (const float*)d_in[3];
  const float* b2  = (const float*)d_in[4];
  const float* W3  = (const float*)d_in[5];
  const float* b3  = (const float*)d_in[6];
  const float* emb = (const float*)d_in[7];

  uint16_t* W1t = (uint16_t*)d_ws;          // 512*1024
  uint16_t* W2t = W1t + 512 * 1024;         // 256*512
  uint16_t* W3t = W2t + 256 * 512;          // 128*256
  uint16_t* EMB = W3t + 128 * 256;          // 1024*128 (padded, normalized)

  prep_weights<<<(688128 + 255) / 256, 256, 0, stream>>>(W1, W2, W3,
                                                         W1t, W2t, W3t);
  prep_emb<<<4, 256, 0, stream>>>(emb, EMB);
  scene_main<<<65536 / MROWS, 128, 0, stream>>>(X, b1, b2, b3, W1t, W2t, W3t,
                                                EMB, (int*)d_out);
}
